// MPNODE_STL10_30047591203083
// MI455X (gfx1250) — matrix operand order repaired and checked by reading, not compiled, not run on hardware
//
#include <hip/hip_runtime.h>
#include <hip/hip_bf16.h>

// ---------------------------------------------------------------------------
// Types for CDNA5 WMMA (gfx1250, wave32)
// ---------------------------------------------------------------------------
typedef __attribute__((ext_vector_type(16))) __bf16 v16bf;
typedef __attribute__((ext_vector_type(8)))  float  v8f;

__device__ __forceinline__ __bf16 f2bf(float f) {
  // round-to-nearest-even f32 -> bf16 via bit manipulation
  union { float f; unsigned int u; } v; v.f = f;
  unsigned int r = v.u + 0x7FFFu + ((v.u >> 16) & 1u);
  unsigned short h = (unsigned short)(r >> 16);
  union { unsigned short s; __bf16 b; } o; o.s = h;
  return o.b;
}

// ---------------------------------------------------------------------------
// Implicit-GEMM conv (3x3 pad1 or 1x1 pad0) with V_WMMA_F32_16X16X32_BF16.
// One wave computes a 64(Cout) x 16(spatial) tile: FOUR WMMAs per K-chunk
// share one B fragment (input gather amortized over 4x the FLOPs).
// K-loop = taps x Cin/32. A (weights) pre-packed in per-lane fragment order.
// flags: 1=+bias, 2=relu, 4=negate.  Requires Cout % 64 == 0.
// ---------------------------------------------------------------------------
__global__ void conv_wmma_kernel(const __bf16* __restrict__ x,
                                 const __bf16* __restrict__ apack,
                                 const float* __restrict__ bias,
                                 float* __restrict__ out,
                                 int N, int Cin, int Cout, int H, int W,
                                 int KHW, int pad, int flags) {
  const int lane   = threadIdx.x & 31;
  const int waveId = threadIdx.x >> 5;
  const int HW     = H * W;
  const int NP     = N * HW;          // spatial GEMM-N (multiple of 16)
  const int ntiles = NP >> 4;
  const int mquads = Cout >> 6;       // 64 output channels per wave
  int tile = blockIdx.x * (int)(blockDim.x >> 5) + waveId;
  const int last = ntiles * mquads - 1;
  tile = tile > last ? last : tile;   // clamp: keeps EXEC all-1s for WMMA
  const int mquad = tile / ntiles;
  const int ntile = tile - mquad * ntiles;

  const int col = lane & 15;          // this lane's B column (spatial pos)
  const int g   = lane >> 4;          // half-wave group
  const int pos = ntile * 16 + col;
  const int n   = pos / HW;
  const int hw  = pos - n * HW;
  const int h   = hw / W;
  const int w   = hw - h * W;

  const int nchunk = Cin >> 5;        // K chunks of 32
  const int K2     = KHW * KHW;
  const size_t mstride = (size_t)K2 * nchunk * 512;  // one 16-row m-tile of A

  v8f acc0 = {0.f, 0.f, 0.f, 0.f, 0.f, 0.f, 0.f, 0.f};
  v8f acc1 = {0.f, 0.f, 0.f, 0.f, 0.f, 0.f, 0.f, 0.f};
  v8f acc2 = {0.f, 0.f, 0.f, 0.f, 0.f, 0.f, 0.f, 0.f};
  v8f acc3 = {0.f, 0.f, 0.f, 0.f, 0.f, 0.f, 0.f, 0.f};
  const __bf16* abase = apack + (size_t)(4 * mquad) * mstride + lane * 16;
  const __bf16  zero  = f2bf(0.f);

  for (int kp = 0; kp < K2; ++kp) {
    const int dh = kp / KHW - pad;
    const int dw = kp - (kp / KHW) * KHW - pad;
    const int hh = h + dh, ww = w + dw;
    const bool valid = (hh >= 0) && (hh < H) && (ww >= 0) && (ww < W);
    const __bf16* xb = x + (long long)n * Cin * HW + (long long)hh * W + ww;
    for (int c = 0; c < nchunk; ++c) {
      const size_t ao = (size_t)(kp * nchunk + c) * 512;
      v16bf b;
      if (valid) {
        const __bf16* xc = xb + (long long)(c * 32 + g * 16) * HW;
#pragma unroll
        for (int e = 0; e < 16; ++e) b[e] = xc[(long long)e * HW];
      } else {
#pragma unroll
        for (int e = 0; e < 16; ++e) b[e] = zero;
      }
      v16bf a0 = *(const v16bf*)(abase + ao);
      v16bf a1 = *(const v16bf*)(abase + mstride + ao);
      v16bf a2 = *(const v16bf*)(abase + 2 * mstride + ao);
      v16bf a3 = *(const v16bf*)(abase + 3 * mstride + ao);
      acc0 = __builtin_amdgcn_wmma_f32_16x16x32_bf16(
          false, a0, false, b, (short)0, acc0, false, false);
      acc1 = __builtin_amdgcn_wmma_f32_16x16x32_bf16(
          false, a1, false, b, (short)0, acc1, false, false);
      acc2 = __builtin_amdgcn_wmma_f32_16x16x32_bf16(
          false, a2, false, b, (short)0, acc2, false, false);
      acc3 = __builtin_amdgcn_wmma_f32_16x16x32_bf16(
          false, a3, false, b, (short)0, acc3, false, false);
    }
  }

  float* ob = out + (size_t)n * Cout * HW + (size_t)h * W + w;
  const int cobase = mquad * 64 + 8 * g;   // C/D layout: VGPR r -> M=r+8g
#pragma unroll
  for (int q = 0; q < 4; ++q) {
    const v8f acc = (q == 0) ? acc0 : (q == 1) ? acc1 : (q == 2) ? acc2 : acc3;
#pragma unroll
    for (int r = 0; r < 8; ++r) {
      const int co = cobase + 16 * q + r;
      float v = acc[r];
      if (flags & 1) v += bias[co];
      if (flags & 2) v = fmaxf(v, 0.f);
      if (flags & 4) v = -v;
      ob[(size_t)co * HW] = v;
    }
  }
}

// ---------------------------------------------------------------------------
// Pack f32 weights into per-lane WMMA A-fragment order (bf16).
// Layout: [mtile][kpos][chunk][lane][16 halves]; lane holds row M=mtile*16+
// (lane&15); element e -> K = chunk*32 + 16g + e.
// transpose=1: effective W'[m][k][kh][kw] = src[k][m][KHW-1-kh][KHW-1-kw]
// (ConvTranspose2d with same weight). k >= CinSrc packs zero (stem padding).
// ---------------------------------------------------------------------------
__global__ void pack_weights_kernel(const float* __restrict__ src,
                                    __bf16* __restrict__ dst,
                                    int CoutE, int CinSrc, int CinPad,
                                    int KHW, int transpose, int total) {
  int idx = blockIdx.x * blockDim.x + threadIdx.x;
  if (idx >= total) return;
  const int e    = idx & 15;
  const int lane = (idx >> 4) & 31;
  int rest = idx >> 9;
  const int nchunk = CinPad >> 5;
  const int K2 = KHW * KHW;
  const int chunk = rest % nchunk; rest /= nchunk;
  const int kp = rest % K2;
  const int mtile = rest / K2;
  const int g = lane >> 4;
  const int m = mtile * 16 + (lane & 15);
  const int kl = 16 * g + e;
  const int k = chunk * 32 + kl;
  const int kh = kp / KHW, kw = kp - (kp / KHW) * KHW;
  float v = 0.f;
  if (k < CinSrc) {
    if (transpose)
      v = src[(((size_t)k * CoutE + m) * KHW + (KHW - 1 - kh)) * KHW + (KHW - 1 - kw)];
    else
      v = src[(((size_t)m * CinSrc + k) * KHW + kh) * KHW + kw];
  }
  dst[idx] = f2bf(v);
}

// ---------------------------------------------------------------------------
// Instance-norm statistics: one block per (n,c), reduce mean / rstd over HW.
// (Also used as global average pool before the head: meanBuf == pooled.)
// ---------------------------------------------------------------------------
__global__ void instnorm_stats_kernel(const float* __restrict__ x,
                                      float* __restrict__ mean,
                                      float* __restrict__ rstd, int HW) {
  const int nc = blockIdx.x;
  const float* p = x + (size_t)nc * HW;
  float s = 0.f, ss = 0.f;
  for (int i = threadIdx.x; i < HW; i += blockDim.x) {
    float v = p[i]; s += v; ss += v * v;
  }
  __shared__ float shs[256], shq[256];
  shs[threadIdx.x] = s; shq[threadIdx.x] = ss;
  __syncthreads();
  for (int o = 128; o > 0; o >>= 1) {
    if ((int)threadIdx.x < o) {
      shs[threadIdx.x] += shs[threadIdx.x + o];
      shq[threadIdx.x] += shq[threadIdx.x + o];
    }
    __syncthreads();
  }
  if (threadIdx.x == 0) {
    float m = shs[0] / (float)HW;
    float var = shq[0] / (float)HW - m * m;
    mean[nc] = m;
    rstd[nc] = rsqrtf(var + 1e-5f);
  }
}

// normalize (no affine) straight into bf16 -- feeds the second conv of ode_rhs
__global__ void normalize_bf16_kernel(const float* __restrict__ x,
                                      const float* __restrict__ mean,
                                      const float* __restrict__ rstd,
                                      __bf16* __restrict__ out,
                                      int HW, int total) {
  int i = blockIdx.x * blockDim.x + threadIdx.x;
  if (i >= total) return;
  const int nc = i / HW;
  out[i] = f2bf((x[i] - mean[nc]) * rstd[nc]);
}

// normalize with affine + relu, f32 out (stem/connector norms)
__global__ void normalize_affine_relu_kernel(const float* __restrict__ x,
                                             const float* __restrict__ mean,
                                             const float* __restrict__ rstd,
                                             const float* __restrict__ gam,
                                             const float* __restrict__ bet,
                                             float* __restrict__ out,
                                             int HW, int C, int total) {
  int i = blockIdx.x * blockDim.x + threadIdx.x;
  if (i >= total) return;
  const int nc = i / HW;
  const int c = nc % C;
  float v = (x[i] - mean[nc]) * rstd[nc] * gam[c] + bet[c];
  out[i] = fmaxf(v, 0.f);
}

// f32 NCHW -> bf16 NCHW with optional channel zero-padding (Cdst >= Csrc;
// padded channels must be pre-zeroed via memset)
__global__ void convert_pad_kernel(const float* __restrict__ src,
                                   __bf16* __restrict__ dst,
                                   int Csrc, int Cdst, int HW, int total) {
  int i = blockIdx.x * blockDim.x + threadIdx.x;
  if (i >= total) return;
  const int hw = i % HW;
  const int t = i / HW;
  const int c = t % Csrc;
  const int n = t / Csrc;
  dst[((size_t)n * Cdst + c) * HW + hw] = f2bf(src[i]);
}

// stage input for next RK4 stage: out_bf16 = bf16(y + a*k)
__global__ void axpy_bf16_kernel(const float* __restrict__ y,
                                 const float* __restrict__ k,
                                 __bf16* __restrict__ out,
                                 float a, int total) {
  int i = blockIdx.x * blockDim.x + threadIdx.x;
  if (i >= total) return;
  out[i] = f2bf(y[i] + a * k[i]);
}

// y += s*(k1 + 2k2 + 2k3 + k4)
__global__ void rk4_combine_kernel(float* __restrict__ y,
                                   const float* __restrict__ k1,
                                   const float* __restrict__ k2,
                                   const float* __restrict__ k3,
                                   const float* __restrict__ k4,
                                   float s, int total) {
  int i = blockIdx.x * blockDim.x + threadIdx.x;
  if (i >= total) return;
  y[i] += s * (k1[i] + 2.f * (k2[i] + k3[i]) + k4[i]);
}

__global__ void avgpool2_kernel(const float* __restrict__ x,
                                float* __restrict__ out,
                                int H, int W, int total) {
  int i = blockIdx.x * blockDim.x + threadIdx.x;
  if (i >= total) return;
  const int W2 = W >> 1, H2 = H >> 1;
  const int w2 = i % W2;
  int t = i / W2;
  const int h2 = t % H2;
  const int nc = t / H2;
  const float* p = x + ((size_t)nc * H + h2 * 2) * W + w2 * 2;
  out[i] = 0.25f * (p[0] + p[1] + p[W] + p[W + 1]);
}

// tiny head: out[n][o] = hb[o] + sum_c pooled[n*C+c] * hw[o*C+c]
__global__ void head_kernel(const float* __restrict__ pooled,
                            const float* __restrict__ hw,
                            const float* __restrict__ hb,
                            float* __restrict__ out, int N, int C, int O) {
  int i = blockIdx.x * blockDim.x + threadIdx.x;
  if (i >= N * O) return;
  const int n = i / O, o = i % O;
  float s = hb[o];
  for (int c = 0; c < C; ++c) s += pooled[(size_t)n * C + c] * hw[(size_t)o * C + c];
  out[i] = s;
}

// ---------------------------------------------------------------------------
// Host orchestration
// ---------------------------------------------------------------------------
extern "C" void kernel_launch(void* const* d_in, const int* in_sizes, int n_in,
                              void* d_out, int out_size, void* d_ws, size_t ws_size,
                              hipStream_t stream) {
  (void)in_sizes; (void)n_in; (void)out_size; (void)ws_size;
  const float* x       = (const float*)d_in[0];
  const float* stem_w  = (const float*)d_in[1];
  const float* stem_b  = (const float*)d_in[2];
  const float* norm1_g = (const float*)d_in[3];
  const float* norm1_b = (const float*)d_in[4];
  const float* ode1_W  = (const float*)d_in[5];
  const float* ode1_b  = (const float*)d_in[6];
  const float* conn1_w = (const float*)d_in[7];
  const float* conn1_b = (const float*)d_in[8];
  const float* norm3_g = (const float*)d_in[9];
  const float* norm3_b = (const float*)d_in[10];
  const float* ode2_W  = (const float*)d_in[11];
  const float* ode2_b  = (const float*)d_in[12];
  const float* conn2_w = (const float*)d_in[13];
  const float* conn2_b = (const float*)d_in[14];
  const float* norm4_g = (const float*)d_in[15];
  const float* norm4_b = (const float*)d_in[16];
  const float* ode3_W  = (const float*)d_in[17];
  const float* ode3_b  = (const float*)d_in[18];
  const float* head_w  = (const float*)d_in[19];
  const float* head_b  = (const float*)d_in[20];
  float* outp = (float*)d_out;

  // ---- carve workspace -----------------------------------------------------
  char* p = (char*)d_ws;
  auto carve = [&](size_t bytes) {
    char* r = p; p += (bytes + 255) & ~(size_t)255; return r;
  };
  const size_t E0 = 4718592;  // 8*64*96*96  (== 8*256*48*48 == 8*1024*24*24)
  const size_t E1 = 9437184;  // 8*128*96*96 (largest f32 tensor, conn1 out)
  float* bufY  = (float*)carve(E0 * 4);
  float* bufZ  = (float*)carve(E1 * 4);
  float* bufK1 = (float*)carve(E1 * 4);
  float* bufK2 = (float*)carve(E0 * 4);
  float* bufK3 = (float*)carve(E0 * 4);
  float* bufK4 = (float*)carve(E0 * 4);
  __bf16* bfA  = (__bf16*)carve(E0 * 2);
  __bf16* bfB  = (__bf16*)carve(E0 * 2);
  float* meanBuf = (float*)carve(4096 * 4);
  float* rstdBuf = (float*)carve(4096 * 4);
  __bf16* packRegion = (__bf16*)carve((size_t)6400000 * 2);

  __bf16* pk = packRegion;
  auto sub = [&](size_t halves) { __bf16* r = pk; pk += halves; return r; };
  __bf16* packStem  = sub(18432);   // (64/16)*9*(32/32)*512
  __bf16* packConn1 = sub(8192);    // (128/16)*1*(64/32)*512
  __bf16* packConn2 = sub(32768);   // (256/16)*1*(128/32)*512
  __bf16 *packO1f[4], *packO1t[4], *packO2f[4], *packO2t[4], *packO3f[4], *packO3t[4];
  for (int i = 0; i < 4; ++i) { packO1f[i] = sub(36864);  packO1t[i] = sub(36864); }
  for (int i = 0; i < 4; ++i) { packO2f[i] = sub(147456); packO2t[i] = sub(147456); }
  for (int i = 0; i < 4; ++i) { packO3f[i] = sub(589824); packO3t[i] = sub(589824); }

  // ---- launch helpers ------------------------------------------------------
  auto launchPack = [&](const float* src, __bf16* dst, int CoutE, int CinSrc,
                        int CinPad, int KHW, int tr) {
    int total = (CoutE / 16) * KHW * KHW * (CinPad / 32) * 512;
    pack_weights_kernel<<<(total + 255) / 256, 256, 0, stream>>>(
        src, dst, CoutE, CinSrc, CinPad, KHW, tr, total);
  };
  auto conv = [&](const __bf16* xin, const __bf16* ap, const float* bias,
                  float* op, int N, int Cin, int Cout, int H, int W,
                  int KHW, int pad, int flags) {
    int tiles = (N * H * W / 16) * (Cout / 64);   // 64 Cout per wave now
    int blocks = (tiles + 3) / 4;   // 4 waves (128 threads) per block
    conv_wmma_kernel<<<blocks, 128, 0, stream>>>(xin, ap, bias, op, N, Cin,
                                                 Cout, H, W, KHW, pad, flags);
  };
  auto stats = [&](const float* xin, int NC, int HW) {
    instnorm_stats_kernel<<<NC, 256, 0, stream>>>(xin, meanBuf, rstdBuf, HW);
  };
  auto normBf = [&](const float* xin, __bf16* op, int HW, int total) {
    normalize_bf16_kernel<<<(total + 255) / 256, 256, 0, stream>>>(
        xin, meanBuf, rstdBuf, op, HW, total);
  };
  auto normAffRelu = [&](const float* xin, const float* g, const float* b,
                         float* op, int HW, int C, int total) {
    normalize_affine_relu_kernel<<<(total + 255) / 256, 256, 0, stream>>>(
        xin, meanBuf, rstdBuf, g, b, op, HW, C, total);
  };
  auto convert = [&](const float* s, __bf16* dst, int N, int Cs, int Cd, int HW) {
    int total = N * Cs * HW;
    convert_pad_kernel<<<(total + 255) / 256, 256, 0, stream>>>(s, dst, Cs, Cd,
                                                                HW, total);
  };
  auto axpyBf = [&](const float* y, const float* k, __bf16* o, float a, int total) {
    axpy_bf16_kernel<<<(total + 255) / 256, 256, 0, stream>>>(y, k, o, a, total);
  };
  auto pool2 = [&](const float* s, float* d, int H, int W, int total) {
    avgpool2_kernel<<<(total + 255) / 256, 256, 0, stream>>>(s, d, H, W, total);
  };

  auto odeRhs = [&](const __bf16* ybf, const __bf16* Wf, const __bf16* Wt,
                    const float* bb, float* kout, int N, int C, int H, int W) {
    const int HW = H * W, total = N * C * HW;
    conv(ybf, Wf, bb, bufZ, N, C, C, H, W, 3, 1, /*bias|relu*/ 3);
    stats(bufZ, N * C, HW);
    normBf(bufZ, bfB, HW, total);
    conv(bfB, Wt, nullptr, kout, N, C, C, H, W, 3, 1, /*negate*/ 4);
  };
  auto odeBlock = [&](float* y, __bf16* const* Wf, __bf16* const* Wt,
                      const float* bbank, int N, int C, int H, int W) {
    const float dtv = 0.25f;
    const int total = N * C * H * W;
    for (int i = 0; i < 4; ++i) {
      const int j = (i + 1 < 3) ? i + 1 : 3;   // min(i+1, 3)
      convert(y, bfA, N, C, C, H * W);
      odeRhs(bfA, Wf[i], Wt[i], bbank + (size_t)i * C, bufK1, N, C, H, W);
      axpyBf(y, bufK1, bfA, 0.5f * dtv, total);
      odeRhs(bfA, Wf[i], Wt[i], bbank + (size_t)i * C, bufK2, N, C, H, W);
      axpyBf(y, bufK2, bfA, 0.5f * dtv, total);
      odeRhs(bfA, Wf[i], Wt[i], bbank + (size_t)i * C, bufK3, N, C, H, W);
      axpyBf(y, bufK3, bfA, dtv, total);
      odeRhs(bfA, Wf[j], Wt[j], bbank + (size_t)j * C, bufK4, N, C, H, W);
      rk4_combine_kernel<<<(total + 255) / 256, 256, 0, stream>>>(
          y, bufK1, bufK2, bufK3, bufK4, dtv / 6.f, total);
    }
  };

  // ---- pack all weight banks (deterministic, every call) -------------------
  launchPack(stem_w, packStem, 64, 3, 32, 3, 0);
  launchPack(conn1_w, packConn1, 128, 64, 64, 1, 0);
  launchPack(conn2_w, packConn2, 256, 128, 128, 1, 0);
  for (int i = 0; i < 4; ++i) {
    launchPack(ode1_W + (size_t)i * 64 * 64 * 9,   packO1f[i], 64, 64, 64, 3, 0);
    launchPack(ode1_W + (size_t)i * 64 * 64 * 9,   packO1t[i], 64, 64, 64, 3, 1);
    launchPack(ode2_W + (size_t)i * 128 * 128 * 9, packO2f[i], 128, 128, 128, 3, 0);
    launchPack(ode2_W + (size_t)i * 128 * 128 * 9, packO2t[i], 128, 128, 128, 3, 1);
    launchPack(ode3_W + (size_t)i * 256 * 256 * 9, packO3f[i], 256, 256, 256, 3, 0);
    launchPack(ode3_W + (size_t)i * 256 * 256 * 9, packO3t[i], 256, 256, 256, 3, 1);
  }

  // ---- stem: conv3x3 (Cin 3->pad 32) + bias, instnorm affine + relu --------
  hipMemsetAsync(bfB, 0, (size_t)8 * 32 * 9216 * 2, stream);
  convert(x, bfB, 8, 3, 32, 9216);
  conv(bfB, packStem, stem_b, bufZ, 8, 32, 64, 96, 96, 3, 1, /*bias*/ 1);
  stats(bufZ, 8 * 64, 9216);
  normAffRelu(bufZ, norm1_g, norm1_b, bufY, 9216, 64, 8 * 64 * 9216);

  // ---- ODE block 1: C=64 @ 96x96 -------------------------------------------
  odeBlock(bufY, packO1f, packO1t, ode1_b, 8, 64, 96, 96);

  // ---- conn1 1x1 (64->128) + instnorm affine + relu + avgpool2 -------------
  convert(bufY, bfA, 8, 64, 64, 9216);
  conv(bfA, packConn1, conn1_b, bufZ, 8, 64, 128, 96, 96, 1, 0, 1);
  stats(bufZ, 8 * 128, 9216);
  normAffRelu(bufZ, norm3_g, norm3_b, bufK1, 9216, 128, 8 * 128 * 9216);
  pool2(bufK1, bufY, 96, 96, 8 * 128 * 48 * 48);

  // ---- ODE block 2: C=128 @ 48x48 ------------------------------------------
  odeBlock(bufY, packO2f, packO2t, ode2_b, 8, 128, 48, 48);

  // ---- conn2 1x1 (128->256) + instnorm affine + relu + avgpool2 ------------
  convert(bufY, bfA, 8, 128, 128, 2304);
  conv(bfA, packConn2, conn2_b, bufZ, 8, 128, 256, 48, 48, 1, 0, 1);
  stats(bufZ, 8 * 256, 2304);
  normAffRelu(bufZ, norm4_g, norm4_b, bufK1, 2304, 256, 8 * 256 * 2304);
  pool2(bufK1, bufY, 48, 48, 8 * 256 * 24 * 24);

  // ---- ODE block 3: C=256 @ 24x24 ------------------------------------------
  odeBlock(bufY, packO3f, packO3t, ode3_b, 8, 256, 24, 24);

  // ---- global average pool (reuse stats: meanBuf == pooled) + head ---------
  stats(bufY, 8 * 256, 576);
  head_kernel<<<1, 128, 0, stream>>>(meanBuf, head_w, head_b, outp, 8, 256, 10);
}
